// EdgeKeyValueNetwork_47571057770947
// MI455X (gfx1250) — matrix operand order repaired and checked by reading, not compiled, not run on hardware
//
#include <hip/hip_runtime.h>

typedef __attribute__((ext_vector_type(16))) _Float16 v16h;
typedef __attribute__((ext_vector_type(8)))  _Float16 h8;
typedef __attribute__((ext_vector_type(8)))  float    v8f;

#define E_TOTAL 131072
#define C3f   0.57735026918962576f   /* 1/sqrt(3) */
#define PW0f  0.20412414523193150f   /* sqrt(1/24) */
#define PW1f  0.35355339059327373f   /* sqrt(3/24) */

// Packed-fragment buffer offsets in d_ws (units of _Float16):
#define P1K_OFF 0
#define P2K_OFF 2048
#define P3K_OFF 6144
#define P1V_OFF 30720
#define P2V_OFF 32768
#define P3V_OFF 36864
#define PACK_HALVES 73728u

__device__ __forceinline__ float silu_f(float x) {
  return x * __builtin_amdgcn_rcpf(1.0f + __expf(-x));
}

__device__ __forceinline__ v8f wmma_f16(v16h a, v16h b, v8f c) {
  return __builtin_amdgcn_wmma_f32_16x16x32_f16(false, a, false, b, (short)0, c, false, false);
}

__device__ __forceinline__ h8 zero8() {
  h8 r;
#pragma unroll
  for (int j = 0; j < 8; ++j) r[j] = (_Float16)0.0f;
  return r;
}
__device__ __forceinline__ h8 splat8(float x) {
  _Float16 hx = (_Float16)x;
  h8 r;
#pragma unroll
  for (int j = 0; j < 8; ++j) r[j] = hx;
  return r;
}
__device__ __forceinline__ h8 ld8(const _Float16* p) { return *(const h8*)p; }

// ---- B fragment loaders -----------------------------------------------------
__device__ __forceinline__ v16h load_B_packed(const _Float16* __restrict__ P, int frag, int lane) {
  return *(const v16h*)(P + (((size_t)frag * 32 + lane) << 4));
}

template<int LD>
__device__ __forceinline__ v16h load_B_gather(const float* __restrict__ W, int kbase, int col) {
  v16h b;
#pragma unroll
  for (int i = 0; i < 16; ++i) b[i] = (_Float16)W[(kbase + i) * LD + col];
  return b;
}

// ---- A fragment from LDS hbuf[32][64] (f16, row-major), 16B ds loads --------
__device__ __forceinline__ v16h load_A_lds(const _Float16* hbuf, int mrow, int hi, int kf) {
  int base = mrow * 64 + kf * 32 + hi * 16;
  h8 lo  = *(const h8*)(hbuf + base);
  h8 hi8 = *(const h8*)(hbuf + base + 8);
  v16h a;
#pragma unroll
  for (int j = 0; j < 8; ++j) { a[j] = lo[j]; a[8 + j] = hi8[j]; }
  return a;
}

// ---- 16->64->64->N3 MLP for 32 edges (two 16-row M tiles) on one wave -------
template<int N3, bool PACKED>
__device__ __forceinline__ void mlp32(
    const float* __restrict__ radial, int e0,
    const float* __restrict__ W1, const float* __restrict__ b1,
    const float* __restrict__ W2, const float* __restrict__ b2,
    const float* __restrict__ W3, const float* __restrict__ b3,
    const _Float16* __restrict__ P1, const _Float16* __restrict__ P2,
    const _Float16* __restrict__ P3,
    _Float16* hbuf, _Float16* wbuf, int row, int hi, int lane)
{
  // layer 1: (32x16)@(16x64), K padded 16->32
  v16h a1[2];
#pragma unroll
  for (int s = 0; s < 2; ++s) {
    const float* rp = radial + (size_t)(e0 + s * 16 + row) * 16;
#pragma unroll
    for (int j = 0; j < 16; ++j) a1[s][j] = (hi == 0) ? (_Float16)rp[j] : (_Float16)0.0f;
  }
#pragma unroll
  for (int nt = 0; nt < 4; ++nt) {
    int c = nt * 16 + row;
    float bb = b1[c];
    v16h bfrag;
    if (PACKED) {
      bfrag = load_B_packed(P1, nt, lane);
    } else {
      if (hi == 0) bfrag = load_B_gather<64>(W1, 0, c);
      else {
#pragma unroll
        for (int i = 0; i < 16; ++i) bfrag[i] = (_Float16)0.0f;
      }
    }
#pragma unroll
    for (int s = 0; s < 2; ++s) {
      v8f acc;
#pragma unroll
      for (int r = 0; r < 8; ++r) acc[r] = bb;
      acc = wmma_f16(a1[s], bfrag, acc);
#pragma unroll
      for (int r = 0; r < 8; ++r)
        hbuf[(s * 16 + r + 8 * hi) * 64 + c] = (_Float16)silu_f(acc[r]);
    }
  }

  // layer 2: (32x64)@(64x64)
  v16h a2[2][2];
#pragma unroll
  for (int s = 0; s < 2; ++s) {
    a2[s][0] = load_A_lds(hbuf, s * 16 + row, hi, 0);
    a2[s][1] = load_A_lds(hbuf, s * 16 + row, hi, 1);
  }
#pragma unroll
  for (int nt = 0; nt < 4; ++nt) {
    int c = nt * 16 + row;
    float bb = b2[c];
    v16h b0 = PACKED ? load_B_packed(P2, nt * 2 + 0, lane)
                     : load_B_gather<64>(W2, hi * 16, c);
    v16h b1f = PACKED ? load_B_packed(P2, nt * 2 + 1, lane)
                      : load_B_gather<64>(W2, 32 + hi * 16, c);
#pragma unroll
    for (int s = 0; s < 2; ++s) {
      v8f acc;
#pragma unroll
      for (int r = 0; r < 8; ++r) acc[r] = bb;
      acc = wmma_f16(a2[s][0], b0, acc);
      acc = wmma_f16(a2[s][1], b1f, acc);
#pragma unroll
      for (int r = 0; r < 8; ++r)
        hbuf[(s * 16 + r + 8 * hi) * 64 + c] = (_Float16)silu_f(acc[r]);
    }
  }

  // layer 3: (32x64)@(64 x N3) streamed into wbuf (f16)
  v16h a3[2][2];
#pragma unroll
  for (int s = 0; s < 2; ++s) {
    a3[s][0] = load_A_lds(hbuf, s * 16 + row, hi, 0);
    a3[s][1] = load_A_lds(hbuf, s * 16 + row, hi, 1);
  }
  for (int nt = 0; nt < N3 / 16; ++nt) {
    int c = nt * 16 + row;
    float bb = b3[c];
    v16h b0 = PACKED ? load_B_packed(P3, nt * 2 + 0, lane)
                     : load_B_gather<N3>(W3, hi * 16, c);
    v16h b1f = PACKED ? load_B_packed(P3, nt * 2 + 1, lane)
                      : load_B_gather<N3>(W3, 32 + hi * 16, c);
#pragma unroll
    for (int s = 0; s < 2; ++s) {
      v8f acc;
#pragma unroll
      for (int r = 0; r < 8; ++r) acc[r] = bb;
      acc = wmma_f16(a3[s][0], b0, acc);
      acc = wmma_f16(a3[s][1], b1f, acc);
#pragma unroll
      for (int r = 0; r < 8; ++r)
        wbuf[(s * 16 + r + 8 * hi) * N3 + c] = (_Float16)acc[r];
    }
  }
}

template<bool PACKED>
__global__ __launch_bounds__(32)
void edge_kv_kernel(
    const float* __restrict__ node, const float* __restrict__ esh,
    const float* __restrict__ radial,
    const float* __restrict__ Wk1, const float* __restrict__ bk1,
    const float* __restrict__ Wk2, const float* __restrict__ bk2,
    const float* __restrict__ Wk3, const float* __restrict__ bk3,
    const float* __restrict__ Wv1, const float* __restrict__ bv1,
    const float* __restrict__ Wv2, const float* __restrict__ bv2,
    const float* __restrict__ Wv3, const float* __restrict__ bv3,
    const _Float16* __restrict__ P1k, const _Float16* __restrict__ P2k,
    const _Float16* __restrict__ P3k, const _Float16* __restrict__ P1v,
    const _Float16* __restrict__ P2v, const _Float16* __restrict__ P3v,
    float* __restrict__ kout, float* __restrict__ vout)
{
  __shared__ _Float16 hbuf[32 * 64];    // 4 KB hidden activations (f16)
  __shared__ _Float16 wbuf[32 * 576];   // 36 KB per-edge TP weights (f16)

  const int lane = threadIdx.x;
  const int row  = lane & 15;
  const int hi   = lane >> 4;
  const int e0   = blockIdx.x * 32;
  const int eg   = e0 + lane;           // FCTP: one full edge per lane

  // per-edge geometric features (loaded once, used by both branches)
  const float* x = node + (size_t)eg * 40;
  const float* shp = esh + (size_t)eg * 4;
  float s[16], vv[8][3], vdot[8];
#pragma unroll
  for (int u = 0; u < 16; ++u) s[u] = x[u];
#pragma unroll
  for (int u = 0; u < 8; ++u) {
    vv[u][0] = x[16 + u * 3 + 0];
    vv[u][1] = x[16 + u * 3 + 1];
    vv[u][2] = x[16 + u * 3 + 2];
  }
  float sh0 = shp[0];
  float sh1[3] = { shp[1], shp[2], shp[3] };
#pragma unroll
  for (int u = 0; u < 8; ++u)
    vdot[u] = vv[u][0] * sh1[0] + vv[u][1] * sh1[1] + vv[u][2] * sh1[2];

  // ============ K branch: (E,384) weights; FCTP mul0=8, mul1=8 ============
  mlp32<384, PACKED>(radial, e0, Wk1, bk1, Wk2, bk2, Wk3, bk3,
                     P1k, P2k, P3k, hbuf, wbuf, row, hi, lane);
  {
    // layout: W000 16x8 @0, W011 16x8 @128, W101 8x8 @256, W110 8x8 @320
    const _Float16* w = wbuf + lane * 384;
    h8 t000 = zero8(), t011 = zero8(), t110 = zero8();
    h8 t101x = zero8(), t101y = zero8(), t101z = zero8();
#pragma unroll
    for (int u = 0; u < 16; ++u) {
      h8 sp = splat8(s[u]);                     // v_pk_fma_f16: 2 MACs/instr
      t000 += sp * ld8(w + u * 8);
      t011 += sp * ld8(w + 128 + u * 8);
    }
#pragma unroll
    for (int u = 0; u < 8; ++u) {
      t110 += splat8(vdot[u]) * ld8(w + 320 + u * 8);
      h8 c1 = ld8(w + 256 + u * 8);
      t101x += c1 * splat8(vv[u][0]);
      t101y += c1 * splat8(vv[u][1]);
      t101z += c1 * splat8(vv[u][2]);
    }
    float ko[32];
#pragma unroll
    for (int ww = 0; ww < 8; ++ww) {
      ko[ww] = PW0f * ((float)t000[ww] * sh0 + C3f * (float)t110[ww]);
      float t011f = (float)t011[ww];
      ko[8 + ww * 3 + 0] = PW1f * C3f * (t011f * sh1[0] + sh0 * (float)t101x[ww]);
      ko[8 + ww * 3 + 1] = PW1f * C3f * (t011f * sh1[1] + sh0 * (float)t101y[ww]);
      ko[8 + ww * 3 + 2] = PW1f * C3f * (t011f * sh1[2] + sh0 * (float)t101z[ww]);
    }
#pragma unroll
    for (int q = 0; q < 8; ++q) {
      float4 o = make_float4(ko[q * 4], ko[q * 4 + 1], ko[q * 4 + 2], ko[q * 4 + 3]);
      *(float4*)(kout + (size_t)eg * 32 + q * 4) = o;
    }
  }

  // ============ V branch: (E,576) weights; FCTP mul0=16, mul1=8 ============
  mlp32<576, PACKED>(radial, e0, Wv1, bv1, Wv2, bv2, Wv3, bv3,
                     P1v, P2v, P3v, hbuf, wbuf, row, hi, lane);
  {
    // layout: W000 16x16 @0, W011 16x8 @256, W101 8x8 @384, W110 8x16 @448
    const _Float16* w = wbuf + lane * 576;
    h8 t0a = zero8(), t0b = zero8(), t1a = zero8(), t1b = zero8();
    h8 t011 = zero8(), t101x = zero8(), t101y = zero8(), t101z = zero8();
#pragma unroll
    for (int u = 0; u < 16; ++u) {
      h8 sp = splat8(s[u]);
      t0a  += sp * ld8(w + u * 16);
      t0b  += sp * ld8(w + u * 16 + 8);
      t011 += sp * ld8(w + 256 + u * 8);
    }
#pragma unroll
    for (int u = 0; u < 8; ++u) {
      h8 dp = splat8(vdot[u]);
      t1a += dp * ld8(w + 448 + u * 16);
      t1b += dp * ld8(w + 448 + u * 16 + 8);
      h8 c1 = ld8(w + 384 + u * 8);
      t101x += c1 * splat8(vv[u][0]);
      t101y += c1 * splat8(vv[u][1]);
      t101z += c1 * splat8(vv[u][2]);
    }
    float vo[40];
#pragma unroll
    for (int ww = 0; ww < 8; ++ww) {
      vo[ww]     = PW0f * ((float)t0a[ww] * sh0 + C3f * (float)t1a[ww]);
      vo[8 + ww] = PW0f * ((float)t0b[ww] * sh0 + C3f * (float)t1b[ww]);
      float t011f = (float)t011[ww];
      vo[16 + ww * 3 + 0] = PW1f * C3f * (t011f * sh1[0] + sh0 * (float)t101x[ww]);
      vo[16 + ww * 3 + 1] = PW1f * C3f * (t011f * sh1[1] + sh0 * (float)t101y[ww]);
      vo[16 + ww * 3 + 2] = PW1f * C3f * (t011f * sh1[2] + sh0 * (float)t101z[ww]);
    }
#pragma unroll
    for (int q = 0; q < 10; ++q) {
      float4 o = make_float4(vo[q * 4], vo[q * 4 + 1], vo[q * 4 + 2], vo[q * 4 + 3]);
      *(float4*)(vout + (size_t)eg * 40 + q * 4) = o;
    }
  }
}

// ---- weight pre-pack kernel: f32 row-major -> f16 fragment-ordered ----------
__global__ void pack_w_kernel(const float* __restrict__ W, _Float16* __restrict__ dst,
                              int Keff, int KF, int N) {
  int idx = blockIdx.x * blockDim.x + threadIdx.x;
  int total = (N / 16) * KF * 512;
  if (idx >= total) return;
  int i    = idx & 15;
  int lane = (idx >> 4) & 31;
  int frag = idx >> 9;
  int f    = frag % KF;
  int nt   = frag / KF;
  int k = f * 32 + (lane >> 4) * 16 + i;
  int n = nt * 16 + (lane & 15);
  float v = (k < Keff) ? W[k * N + n] : 0.0f;
  dst[idx] = (_Float16)v;
}

extern "C" void kernel_launch(void* const* d_in, const int* in_sizes, int n_in,
                              void* d_out, int out_size, void* d_ws, size_t ws_size,
                              hipStream_t stream) {
  const float* node = (const float*)d_in[0];
  const float* esh  = (const float*)d_in[1];
  const float* rad  = (const float*)d_in[2];
  const float* Wk1 = (const float*)d_in[3];   const float* bk1 = (const float*)d_in[4];
  const float* Wk2 = (const float*)d_in[5];   const float* bk2 = (const float*)d_in[6];
  const float* Wk3 = (const float*)d_in[7];   const float* bk3 = (const float*)d_in[8];
  const float* Wv1 = (const float*)d_in[9];   const float* bv1 = (const float*)d_in[10];
  const float* Wv2 = (const float*)d_in[11];  const float* bv2 = (const float*)d_in[12];
  const float* Wv3 = (const float*)d_in[13];  const float* bv3 = (const float*)d_in[14];

  float* kout = (float*)d_out;                    // (E, 32)
  float* vout = kout + (size_t)E_TOTAL * 32;      // (E, 40)

  dim3 grid(E_TOTAL / 32);
  dim3 block(32);

  if (ws_size >= (size_t)PACK_HALVES * sizeof(unsigned short)) {
    _Float16* P = (_Float16*)d_ws;
    struct { const float* W; size_t off; int Keff, KF, N; } jobs[6] = {
      { Wk1, P1K_OFF, 16, 1, 64  }, { Wk2, P2K_OFF, 64, 2, 64 },
      { Wk3, P3K_OFF, 64, 2, 384 }, { Wv1, P1V_OFF, 16, 1, 64 },
      { Wv2, P2V_OFF, 64, 2, 64  }, { Wv3, P3V_OFF, 64, 2, 576 },
    };
    for (int j = 0; j < 6; ++j) {
      int total = (jobs[j].N / 16) * jobs[j].KF * 512;
      pack_w_kernel<<<(total + 255) / 256, 256, 0, stream>>>(
          jobs[j].W, P + jobs[j].off, jobs[j].Keff, jobs[j].KF, jobs[j].N);
    }
    edge_kv_kernel<true><<<grid, block, 0, stream>>>(
        node, esh, rad, Wk1, bk1, Wk2, bk2, Wk3, bk3,
        Wv1, bv1, Wv2, bv2, Wv3, bv3,
        P + P1K_OFF, P + P2K_OFF, P + P3K_OFF,
        P + P1V_OFF, P + P2V_OFF, P + P3V_OFF, kout, vout);
  } else {
    edge_kv_kernel<false><<<grid, block, 0, stream>>>(
        node, esh, rad, Wk1, bk1, Wk2, bk2, Wk3, bk3,
        Wv1, bv1, Wv2, bv2, Wv3, bv3,
        nullptr, nullptr, nullptr, nullptr, nullptr, nullptr, kout, vout);
  }
}
